// AttentionBase_43679817400717
// MI455X (gfx1250) — matrix operand order repaired and checked by reading, not compiled, not run on hardware
//
#include <hip/hip_runtime.h>

// ---------------------------------------------------------------------------
// Bilinear attention:  score = softmax(q@W@k^T + mask),  ctx = score @ v
// B=16, TQ=TK=DQ=DK=DV=1024, fp32 I/O, bf16 WMMA internally.
// HBM-bound (~0.6 GB traffic vs ~103 GFLOP) -> bf16 tensor path + async
// global->LDS staging (ASYNCcnt, double buffered) + 32x64 per-wave register
// tile (2 A frags x 4 B frags -> 8 WMMAs per K-step, B frags reused 2x).
// Pipeline:
//   k_qw_gemm   : qW(bf16, ws) = query @ W
//   k_score_gemm: score(f32,out) = qW @ keys^T + mask
//   k_softmax   : row softmax of score in place
//   k_ctx_gemm  : ctx(f32,out) = score @ values
// ---------------------------------------------------------------------------

typedef __attribute__((ext_vector_type(16))) __bf16 v16bf;
typedef __attribute__((ext_vector_type(8)))  float  v8f;

constexpr int NB = 16;     // batch
constexpr int TQ = 1024;
constexpr int TK = 1024;
constexpr int DD = 1024;   // DQ == DK == DV
constexpr int KSTEP = 32;  // K per WMMA

constexpr int LDS_KN = 68; // [k][n] tile: 32 rows x 64 n + 4 pad (f32), 272B row
constexpr int LDS_NK = 36; // [n][k] tile: 64 rows x 32 k + 4 pad (f32), 144B row

// ---- CDNA5 async global->LDS copy (GLOBAL_LOAD_ASYNC_TO_LDS_B128) ---------
// Per-lane: 16 bytes global -> LDS, tracked by ASYNCcnt, bypasses VGPRs.
__device__ __forceinline__ void async_copy_b128(void* lds_ptr, const float* g) {
  // Low 32 bits of a generic pointer to LDS are the LDS byte address.
  unsigned lds = (unsigned)(uintptr_t)lds_ptr;
  asm volatile("global_load_async_to_lds_b128 %0, %1, off"
               :: "v"(lds), "v"(g)
               : "memory");
}

__device__ __forceinline__ void wait_async0() {
#if __has_builtin(__builtin_amdgcn_s_wait_asynccnt)
  __builtin_amdgcn_s_wait_asynccnt(0);
#else
  asm volatile("s_wait_asynccnt 0x0" ::: "memory");
#endif
}

// ---- fragment loaders ------------------------------------------------------

// A fragment 16x32 bf16 from a row-major f32 matrix (convert on load).
// lanes 0-15: row M=lane, K = 0..15; lanes 16-31: row M=lane-16,
// K = 16..31.
__device__ __forceinline__ v16bf load_a_frag_f32(const float* __restrict__ A,
                                                 int row, int k0, int lane) {
  const int kb = (lane < 16) ? 0 : 16;
  const float* p = A + (size_t)row * DD + k0 + kb;
  v16bf a;
#pragma unroll
  for (int j = 0; j < 8; ++j) {
    a[j]     = (__bf16)p[j];
    a[8 + j] = (__bf16)p[8 + j];
  }
  return a;
}

// Same, from a row-major bf16 matrix.
__device__ __forceinline__ v16bf load_a_frag_bf16(const __bf16* __restrict__ A,
                                                  int row, int k0, int lane) {
  const int kb = (lane < 16) ? 0 : 16;
  const __bf16* p = A + (size_t)row * DD + k0 + kb;
  v16bf a;
#pragma unroll
  for (int j = 0; j < 8; ++j) {
    a[j]     = p[j];
    a[8 + j] = p[8 + j];
  }
  return a;
}

// B fragment from LDS f32 tile in [n][k] layout (row stride LDS_NK).
// lane column n = nloc; lanes 0-15 -> K 0..15, lanes 16-31 -> K 16..31.
// 64B contiguous per lane -> ds_load_b128 x4, then cvt to bf16.
__device__ __forceinline__ v16bf load_b_frag_nk(const float* Bs, int nloc,
                                                int lane) {
  const int kb = (lane < 16) ? 0 : 16;
  const float* p = Bs + nloc * LDS_NK + kb;
  v16bf b;
#pragma unroll
  for (int j = 0; j < 16; ++j) b[j] = (__bf16)p[j];
  return b;
}

// B fragment from LDS f32 tile in [k][n] layout (row stride LDS_KN).
__device__ __forceinline__ v16bf load_b_frag_kn(const float* Bs, int nloc,
                                                int lane) {
  const int kb = (lane < 16) ? 0 : 16;
  const float* p = Bs + kb * LDS_KN + nloc;
  v16bf b;
#pragma unroll
  for (int j = 0; j < 16; ++j) b[j] = (__bf16)p[j * LDS_KN];
  return b;
}

// ---- async stagers (2 x b128 per thread per 8KB tile, 256 threads) --------

// Stage src[k0:k0+32, n0:n0+64] (row stride 1024) into [k][n] LDS tile.
__device__ __forceinline__ void stage_kn(float* buf, const float* src,
                                         int tid) {
#pragma unroll
  for (int s = 0; s < 2; ++s) {
    const int idx = tid + s * 256;   // 0..511
    const int r   = idx >> 4;        // 0..31 : k row
    const int c4  = (idx & 15) * 4;  // 0..60 : n
    async_copy_b128(buf + r * LDS_KN + c4, src + (size_t)r * DD + c4);
  }
}

// Stage src[n0:n0+64, k0:k0+32] (row stride 1024) into [n][k] LDS tile.
__device__ __forceinline__ void stage_nk(float* buf, const float* src,
                                         int tid) {
#pragma unroll
  for (int s = 0; s < 2; ++s) {
    const int idx = tid + s * 256;   // 0..511
    const int r   = idx >> 3;        // 0..63 : n row
    const int c4  = (idx & 7) * 4;   // 0..28 : k
    async_copy_b128(buf + r * LDS_NK + c4, src + (size_t)r * DD + c4);
  }
}

// ---- kernel 1: qW = query @ W  ([k][n] async-staged B, 32x64/wave) --------
__global__ void __launch_bounds__(256)
k_qw_gemm(const float* __restrict__ Q, const float* __restrict__ W,
          __bf16* __restrict__ qWbf) {
  __shared__ float Bs[2][32 * LDS_KN];
  const int tid  = threadIdx.x;
  const int lane = tid & 31;
  const int wave = tid >> 5;
  const int m0 = blockIdx.x * 256 + wave * 32;  // rows over B*TQ = 16384
  const int n0 = blockIdx.y * 64;
  const int arow0 = m0 + (lane & 15);
  const int arow1 = arow0 + 16;

  v8f acc0[4] = {}, acc1[4] = {};
  stage_kn(Bs[0], W + n0, tid);                       // tile for kk=0
  constexpr int NT = DD / KSTEP;
  for (int it = 0; it < NT; ++it) {
    const int kk = it * KSTEP;
    wait_async0();
    __syncthreads();
    if (it + 1 < NT)
      stage_kn(Bs[(it + 1) & 1], W + (size_t)(kk + KSTEP) * DD + n0, tid);
    const float* cur = Bs[it & 1];
    v16bf a0 = load_a_frag_f32(Q, arow0, kk, lane);
    v16bf a1 = load_a_frag_f32(Q, arow1, kk, lane);
#pragma unroll
    for (int t = 0; t < 4; ++t) {
      v16bf b = load_b_frag_kn(cur, t * 16 + (lane & 15), lane);
      acc0[t] = __builtin_amdgcn_wmma_f32_16x16x32_bf16(
          false, a0, false, b, (short)0, acc0[t], false, false);
      acc1[t] = __builtin_amdgcn_wmma_f32_16x16x32_bf16(
          false, a1, false, b, (short)0, acc1[t], false, false);
    }
  }
  const int nl = lane & 15;
  const int mh = (lane < 16) ? 0 : 8;
#pragma unroll
  for (int t = 0; t < 4; ++t)
#pragma unroll
    for (int v = 0; v < 8; ++v) {
      qWbf[(size_t)(m0 + mh + v) * DD + n0 + t * 16 + nl] = (__bf16)acc0[t][v];
      qWbf[(size_t)(m0 + 16 + mh + v) * DD + n0 + t * 16 + nl] =
          (__bf16)acc1[t][v];
    }
}

// ---- kernel 2: weight = qW @ keys^T + mask  ([n][k] async-staged keys) ----
__global__ void __launch_bounds__(256)
k_score_gemm(const __bf16* __restrict__ qWbf, const float* __restrict__ Keys,
             const float* __restrict__ mask, float* __restrict__ score) {
  __shared__ float Bs[2][64 * LDS_NK];
  const int tid  = threadIdx.x;
  const int lane = tid & 31;
  const int wave = tid >> 5;
  const int b    = blockIdx.z;
  const int m0 = blockIdx.x * 256 + wave * 32;  // query rows
  const int n0 = blockIdx.y * 64;               // key rows (output cols)
  const __bf16* A  = qWbf + (size_t)b * TQ * DD;
  const float*  Kb = Keys + (size_t)b * TK * DD;
  const int arow0 = m0 + (lane & 15);
  const int arow1 = arow0 + 16;

  v8f acc0[4] = {}, acc1[4] = {};
  stage_nk(Bs[0], Kb + (size_t)n0 * DD, tid);
  constexpr int NT = DD / KSTEP;
  for (int it = 0; it < NT; ++it) {
    const int kk = it * KSTEP;
    wait_async0();
    __syncthreads();
    if (it + 1 < NT)
      stage_nk(Bs[(it + 1) & 1], Kb + (size_t)n0 * DD + kk + KSTEP, tid);
    const float* cur = Bs[it & 1];
    v16bf a0 = load_a_frag_bf16(A, arow0, kk, lane);
    v16bf a1 = load_a_frag_bf16(A, arow1, kk, lane);
#pragma unroll
    for (int t = 0; t < 4; ++t) {
      v16bf bb = load_b_frag_nk(cur, t * 16 + (lane & 15), lane);
      acc0[t] = __builtin_amdgcn_wmma_f32_16x16x32_bf16(
          false, a0, false, bb, (short)0, acc0[t], false, false);
      acc1[t] = __builtin_amdgcn_wmma_f32_16x16x32_bf16(
          false, a1, false, bb, (short)0, acc1[t], false, false);
    }
  }
  const int nl = lane & 15;
  const int mh = (lane < 16) ? 0 : 8;
  float* S = score + (size_t)b * TQ * TK;
#pragma unroll
  for (int t = 0; t < 4; ++t) {
    const int c = n0 + t * 16 + nl;
    const float mv = mask[b * TK + c];
#pragma unroll
    for (int v = 0; v < 8; ++v) {
      S[(size_t)(m0 + mh + v) * TK + c]      = acc0[t][v] + mv;
      S[(size_t)(m0 + 16 + mh + v) * TK + c] = acc1[t][v] + mv;
    }
  }
}

// ---- kernel 3: row softmax in place (one block per row of 1024) -----------
__global__ void __launch_bounds__(256)
k_softmax(float* __restrict__ score) {
  __shared__ float red[256];
  float* p = score + (size_t)blockIdx.x * TK;
  const int tid = threadIdx.x;

  float x[4];
  float mx = -3.402823466e38f;
#pragma unroll
  for (int i = 0; i < 4; ++i) {
    x[i] = p[tid + i * 256];
    mx = fmaxf(mx, x[i]);
  }
  red[tid] = mx;
  __syncthreads();
  for (int s = 128; s > 0; s >>= 1) {
    if (tid < s) red[tid] = fmaxf(red[tid], red[tid + s]);
    __syncthreads();
  }
  mx = red[0];
  __syncthreads();

  float sum = 0.f;
#pragma unroll
  for (int i = 0; i < 4; ++i) {
    x[i] = __expf(x[i] - mx);
    sum += x[i];
  }
  red[tid] = sum;
  __syncthreads();
  for (int s = 128; s > 0; s >>= 1) {
    if (tid < s) red[tid] += red[tid + s];
    __syncthreads();
  }
  const float inv = 1.0f / red[0];
#pragma unroll
  for (int i = 0; i < 4; ++i) p[tid + i * 256] = x[i] * inv;
}

// ---- kernel 4: ctx = score @ values  ([k][n] async-staged values) ---------
__global__ void __launch_bounds__(256)
k_ctx_gemm(const float* __restrict__ score, const float* __restrict__ V,
           float* __restrict__ ctx) {
  __shared__ float Bs[2][32 * LDS_KN];
  const int tid  = threadIdx.x;
  const int lane = tid & 31;
  const int wave = tid >> 5;
  const int b    = blockIdx.z;
  const int m0 = blockIdx.x * 256 + wave * 32;
  const int n0 = blockIdx.y * 64;
  const float* A  = score + (size_t)b * TQ * TK;
  const float* Vb = V     + (size_t)b * TK * DD;
  const int arow0 = m0 + (lane & 15);
  const int arow1 = arow0 + 16;

  v8f acc0[4] = {}, acc1[4] = {};
  stage_kn(Bs[0], Vb + n0, tid);
  constexpr int NT = TK / KSTEP;
  for (int it = 0; it < NT; ++it) {
    const int kk = it * KSTEP;
    wait_async0();
    __syncthreads();
    if (it + 1 < NT)
      stage_kn(Bs[(it + 1) & 1], Vb + (size_t)(kk + KSTEP) * DD + n0, tid);
    const float* cur = Bs[it & 1];
    v16bf a0 = load_a_frag_f32(A, arow0, kk, lane);
    v16bf a1 = load_a_frag_f32(A, arow1, kk, lane);
#pragma unroll
    for (int t = 0; t < 4; ++t) {
      v16bf bb = load_b_frag_kn(cur, t * 16 + (lane & 15), lane);
      acc0[t] = __builtin_amdgcn_wmma_f32_16x16x32_bf16(
          false, a0, false, bb, (short)0, acc0[t], false, false);
      acc1[t] = __builtin_amdgcn_wmma_f32_16x16x32_bf16(
          false, a1, false, bb, (short)0, acc1[t], false, false);
    }
  }
  const int nl = lane & 15;
  const int mh = (lane < 16) ? 0 : 8;
  float* C = ctx + (size_t)b * TQ * DD;
#pragma unroll
  for (int t = 0; t < 4; ++t)
#pragma unroll
    for (int v = 0; v < 8; ++v) {
      C[(size_t)(m0 + mh + v) * DD + n0 + t * 16 + nl]      = acc0[t][v];
      C[(size_t)(m0 + 16 + mh + v) * DD + n0 + t * 16 + nl] = acc1[t][v];
    }
}

// ---------------------------------------------------------------------------
extern "C" void kernel_launch(void* const* d_in, const int* in_sizes, int n_in,
                              void* d_out, int out_size, void* d_ws,
                              size_t ws_size, hipStream_t stream) {
  const float* query  = (const float*)d_in[0];  // [B,TQ,DQ]
  const float* keys   = (const float*)d_in[1];  // [B,TK,DK]
  const float* values = (const float*)d_in[2];  // [B,TK,DV]
  const float* W      = (const float*)d_in[3];  // [DQ,DK]
  const float* mask   = (const float*)d_in[4];  // [B,TK]

  float* score = (float*)d_out;                            // [B,TQ,TK]
  float* ctx   = score + (size_t)NB * TQ * TK;             // [B,TQ,DV]
  __bf16* qWbf = (__bf16*)d_ws;                            // [B,TQ,DK] bf16

  dim3 blk(256);
  // 1) qW = query @ W       (M = B*TQ flattened, 256 rows / block)
  k_qw_gemm<<<dim3((NB * TQ) / 256, DD / 64), blk, 0, stream>>>(query, W, qWbf);
  // 2) score = qW @ keys^T + mask
  k_score_gemm<<<dim3(TQ / 256, TK / 64, NB), blk, 0, stream>>>(qWbf, keys,
                                                                mask, score);
  // 3) softmax rows in place
  k_softmax<<<dim3(NB * TQ), blk, 0, stream>>>(score);
  // 4) ctx = score @ values
  k_ctx_gemm<<<dim3(TQ / 256, DD / 64, NB), blk, 0, stream>>>(score, values,
                                                              ctx);
}
